// SparseMultiheadAttention_58231166599227
// MI455X (gfx1250) — matrix operand order repaired and checked by reading, not compiled, not run on hardware
//
#include <hip/hip_runtime.h>

#define DM    1024
#define NH    16
#define DK    64
#define BATCH 8
#define SEQ   1024

typedef __attribute__((ext_vector_type(16))) __bf16 bf16x16;
typedef __attribute__((ext_vector_type(8)))  __bf16 bf16x8;
typedef __attribute__((ext_vector_type(8)))  float  f32x8;
typedef __attribute__((ext_vector_type(4)))  float  f32x4;
typedef __attribute__((ext_vector_type(4)))  unsigned int u32x4;
typedef __attribute__((ext_vector_type(8)))  int i32x8;
typedef __attribute__((ext_vector_type(4)))  int i32x4;

#if __has_builtin(__builtin_amdgcn_tensor_load_to_lds) && __has_builtin(__builtin_amdgcn_s_wait_tensorcnt)
#define USE_TDM 1
#else
#define USE_TDM 0
#endif

#if __has_builtin(__builtin_amdgcn_update_dpp)
#define USE_DPP 1
#else
#define USE_DPP 0
#endif

// ---------------------------------------------------------------------------
// WMMA wrapper: D = A(16x32 bf16) * B(32x16 bf16) + C(16x16 f32)
// ---------------------------------------------------------------------------
__device__ __forceinline__ f32x8 wmma_bf16(bf16x16 a, bf16x16 b, f32x8 c) {
    return __builtin_amdgcn_wmma_f32_16x16x32_bf16(
        false, a, false, b, (short)0, c, false, false);
}

__device__ __forceinline__ bf16x16 cat8(bf16x8 lo, bf16x8 hi) {
    bf16x16 r;
#pragma unroll
    for (int i = 0; i < 8; ++i) { r[i] = lo[i]; r[8 + i] = hi[i]; }
    return r;
}

// A-fragment (16x32, bf16) from fp32 row-major source (src pre-offset to tile row 0).
__device__ __forceinline__ bf16x16 load_a_f32(const float* __restrict__ src,
                                              int lda, int k0, int lane) {
    int m = lane & 15, kh = lane >> 4;
    const float* p0 = src + m * lda + k0 + kh * 8;
    f32x4 a0 = *(const f32x4*)(p0);
    f32x4 a1 = *(const f32x4*)(p0 + 4);
    f32x4 a2 = *(const f32x4*)(p0 + 16);
    f32x4 a3 = *(const f32x4*)(p0 + 20);
    bf16x16 r;
#pragma unroll
    for (int i = 0; i < 4; ++i) {
        r[i]      = (__bf16)a0[i];
        r[4 + i]  = (__bf16)a1[i];
        r[8 + i]  = (__bf16)a2[i];
        r[12 + i] = (__bf16)a3[i];
    }
    return r;
}

// A-fragment (16x32) from bf16 row-major source (src pre-offset to tile row 0).
__device__ __forceinline__ bf16x16 load_a_bf16(const __bf16* __restrict__ src,
                                               int lda, int k0, int lane) {
    int m = lane & 15, kh = lane >> 4;
    const __bf16* p0 = src + m * lda + k0 + kh * 8;
    bf16x8 lo = *(const bf16x8*)(p0);
    bf16x8 hi = *(const bf16x8*)(p0 + 16);
    return cat8(lo, hi);
}

// B-fragment (32x16), transposed access: B(k,n) = S[n0+n][k0+k], S row-major ld.
__device__ __forceinline__ bf16x16 load_b_tr(const __bf16* __restrict__ S,
                                             int ld, int n0, int k0, int lane) {
    int n = lane & 15, kh = lane >> 4;
    const __bf16* p = S + (size_t)(n0 + n) * ld + k0 + kh * 8;
    bf16x8 lo = *(const bf16x8*)(p);
    bf16x8 hi = *(const bf16x8*)(p + 16);
    return cat8(lo, hi);
}

// ---------------------------------------------------------------------------
// DPP butterfly reductions over the 16-lane column groups of the C layout.
// ---------------------------------------------------------------------------
#if USE_DPP
template <int CTRL>
__device__ __forceinline__ float dpp_mov_f(float x) {
    int xi = __builtin_bit_cast(int, x);
    int r  = __builtin_amdgcn_update_dpp(xi, xi, CTRL, 0xf, 0xf, false);
    return __builtin_bit_cast(float, r);
}
__device__ __forceinline__ float red16_max(float x) {
    x = fmaxf(x, dpp_mov_f<0x0B1>(x));   // quad_perm [1,0,3,2]  (xor 1)
    x = fmaxf(x, dpp_mov_f<0x04E>(x));   // quad_perm [2,3,0,1]  (xor 2)
    x = fmaxf(x, dpp_mov_f<0x141>(x));   // row_half_mirror      (8-groups)
    x = fmaxf(x, dpp_mov_f<0x140>(x));   // row_mirror           (16-groups)
    return x;
}
__device__ __forceinline__ float red16_sum(float x) {
    x += dpp_mov_f<0x0B1>(x);
    x += dpp_mov_f<0x04E>(x);
    x += dpp_mov_f<0x141>(x);
    x += dpp_mov_f<0x140>(x);
    return x;
}
#else
__device__ __forceinline__ float red16_max(float x) {
    for (int off = 1; off < 16; off <<= 1) x = fmaxf(x, __shfl_xor(x, off, 32));
    return x;
}
__device__ __forceinline__ float red16_sum(float x) {
    for (int off = 1; off < 16; off <<= 1) x += __shfl_xor(x, off, 32);
    return x;
}
#endif

// ---------------------------------------------------------------------------
// Tensor Data Mover: 2D tile (bf16) -> LDS, wave-uniform descriptor in SGPRs.
// amdgpu-toolchain (clang-23) 6-arg form:
//   (u32x4 g0, i32x8 g1, i32x4 g2, i32x4 g3, i32x8 pad, i32 cpol)
// ---------------------------------------------------------------------------
#if USE_TDM
__device__ __forceinline__ unsigned lds_off_of(const void* p) {
    return (unsigned)(unsigned long long)
        (__attribute__((address_space(3))) const void*)p;
}
__device__ __forceinline__ void tdm_load_2d(unsigned lds_off, const void* gaddr,
                                            unsigned tensor_d0, unsigned tensor_d1,
                                            unsigned stride0,
                                            unsigned tile_d0, unsigned tile_d1) {
    unsigned long long ga = (unsigned long long)gaddr;
    u32x4 g0;
    g0[0] = 1u;                                             // count=1 (valid D#)
    g0[1] = lds_off;                                        // lds_addr
    g0[2] = (unsigned)ga;                                   // global_addr[31:0]
    g0[3] = (unsigned)((ga >> 32) & 0x01FFFFFFull)          // global_addr[56:32]
          | (2u << 30);                                     // type=2 (image)
    i32x8 g1;
    g1[0] = (int)(1u << 16);                                // data_size=1 -> 2B
    g1[1] = (int)(tensor_d0 << 16);                         // tensor_dim0 lo16
    g1[2] = (int)((tensor_d0 >> 16) | (tensor_d1 << 16));   // dim0 hi / dim1 lo
    g1[3] = (int)((tensor_d1 >> 16) | (tile_d0 << 16));     // dim1 hi / tile_dim0
    g1[4] = (int)(tile_d1 & 0xffff);                        // tile_dim1; tile_dim2=0
    g1[5] = (int)stride0;                                   // tensor_dim0_stride
    g1[6] = 0;
    g1[7] = 0;
    i32x4 z4 = {0, 0, 0, 0};
    i32x8 z8 = {0, 0, 0, 0, 0, 0, 0, 0};
    __builtin_amdgcn_tensor_load_to_lds(g0, g1, z4, z4, z8, 0);
}
#endif

// ---------------------------------------------------------------------------
// fp32 -> bf16 weight conversion
// ---------------------------------------------------------------------------
__global__ void cvt_f32_bf16_kernel(const float* __restrict__ in,
                                    __bf16* __restrict__ out, int n) {
    int i = (blockIdx.x * blockDim.x + threadIdx.x) * 4;
    if (i >= n) return;
    f32x4 v = *(const f32x4*)(in + i);
    __bf16 o[4];
#pragma unroll
    for (int j = 0; j < 4; ++j) o[j] = (__bf16)v[j];
    *(uint2*)(out + i) = *(uint2*)o;
}

// ---------------------------------------------------------------------------
// Projection GEMM: out = X @ W^T + b. One wave computes a 16x128 strip.
// MODE 0: store [B,H,S,DK] bf16; MODE 1: store [B,H,DK,S] bf16 (transposed V).
// ---------------------------------------------------------------------------
template <int MODE>
__global__ void __launch_bounds__(256)
proj_kernel(const float* __restrict__ X, const __bf16* __restrict__ W,
            const float* __restrict__ bias, __bf16* __restrict__ out) {
    int wid  = (blockIdx.x * blockDim.x + threadIdx.x) >> 5;
    int lane = threadIdx.x & 31;
    const int NSTRIP = DM / 128;                // 8
    int mt = wid / NSTRIP, nt = wid % NSTRIP;
    int row0 = mt * 16, ncol0 = nt * 128;
    const float* A = X + (size_t)row0 * DM;

    f32x8 acc[8] = {};
    for (int k0 = 0; k0 < DM; k0 += 32) {
        if (k0 + 32 < DM) {
            __builtin_prefetch(A + (lane & 15) * DM + k0 + 32, 0, 3);
            __builtin_prefetch(W + (size_t)(ncol0 + (lane & 15)) * DM + k0 + 32, 0, 3);
        }
        bf16x16 a = load_a_f32(A, DM, k0, lane);
#pragma unroll
        for (int t = 0; t < 8; ++t) {
            bf16x16 b = load_b_tr(W, DM, ncol0 + t * 16, k0, lane);
            acc[t] = wmma_bf16(a, b, acc[t]);
        }
    }

    int n_ = lane & 15, kh = lane >> 4;
#pragma unroll
    for (int t = 0; t < 8; ++t) {
        int ncol = ncol0 + t * 16 + n_;
        float bv = bias[ncol];
        int h = ncol >> 6, dd = ncol & 63;
        if (MODE == 0) {                        // [B,H,S,DK]
#pragma unroll
            for (int j = 0; j < 8; ++j) {
                int s = row0 + kh * 8 + j;
                int b = s >> 10, sq = s & 1023;
                out[(((size_t)(b * NH + h) * SEQ) + sq) * DK + dd] =
                    (__bf16)(acc[t][j] + bv);
            }
        } else {                                // [B,H,DK,S]: contiguous 16B store
            int s0 = row0 + kh * 8;
            int b = s0 >> 10, sq0 = s0 & 1023;
            bf16x8 v;
#pragma unroll
            for (int j = 0; j < 8; ++j) v[j] = (__bf16)(acc[t][j] + bv);
            *(bf16x8*)(out + ((size_t)(b * NH + h) * DK + dd) * SEQ + sq0) = v;
        }
    }
}

// ---------------------------------------------------------------------------
// Flash attention. Block = 4 waves sharing one (b,h); each wave one 16-query
// tile. K/V 64x64 tiles double-buffered in LDS via the Tensor Data Mover.
// ---------------------------------------------------------------------------
__global__ void __launch_bounds__(128)
attn_kernel(const __bf16* __restrict__ Qh, const __bf16* __restrict__ Kh,
            const __bf16* __restrict__ Vt, const unsigned char* __restrict__ mask,
            __bf16* __restrict__ Oc) {
#if USE_TDM
    __shared__ __bf16 Kbuf[2][64][64];          // 16 KB
    __shared__ __bf16 Vbuf[2][64][64];          // 16 KB
#endif
    __shared__ __bf16 Plds[4][16][64];          // 8 KB (wave-private strips)

    int wline = threadIdx.x >> 5;
    int lane  = threadIdx.x & 31;
    int blk   = blockIdx.x;                     // 2048 = 8*16*16
    int qg = blk & 15, h = (blk >> 4) & 15, b = blk >> 8;
    int qt = qg * 4 + wline;
    int q0 = qt * 16;

    const __bf16* Qp = Qh + ((size_t)(b * NH + h) * SEQ + q0) * DK;
    const __bf16* Kp = Kh + (size_t)(b * NH + h) * SEQ * DK;
    const __bf16* Vp = Vt + (size_t)(b * NH + h) * DK * SEQ;
    const unsigned char* Mp = mask + ((size_t)(b * NH + h) * SEQ + q0) * SEQ;

    bf16x16 aq0 = load_a_bf16(Qp, DK, 0,  lane);
    bf16x16 aq1 = load_a_bf16(Qp, DK, 32, lane);

    f32x8 acc[4] = {};
    float mrun[8], lrun[8];
#pragma unroll
    for (int j = 0; j < 8; ++j) { mrun[j] = -1e30f; lrun[j] = 0.0f; }

    int n_ = lane & 15, kh = lane >> 4;
    const float scale = 0.125f;                 // 1/sqrt(64)

#if USE_TDM
    if (wline == 0) {
        tdm_load_2d(lds_off_of(&Kbuf[0][0][0]), Kp, DK, SEQ, DK, DK, 64);
        tdm_load_2d(lds_off_of(&Vbuf[0][0][0]), Vp, SEQ, DK, SEQ, 64, DK);
    }
#endif

    for (int kk = 0; kk < SEQ; kk += 64) {
#if USE_TDM
        int cur = (kk >> 6) & 1;
        if (wline == 0) {
            if (kk + 64 < SEQ) {
                int nxt = cur ^ 1;
                tdm_load_2d(lds_off_of(&Kbuf[nxt][0][0]),
                            Kp + (size_t)(kk + 64) * DK, DK, SEQ, DK, DK, 64);
                tdm_load_2d(lds_off_of(&Vbuf[nxt][0][0]),
                            Vp + (kk + 64), SEQ, DK, SEQ, 64, DK);
                __builtin_amdgcn_s_wait_tensorcnt(2);   // prev tile complete
            } else {
                __builtin_amdgcn_s_wait_tensorcnt(0);
            }
        }
        __syncthreads();                        // tile `cur` visible to all waves
        const __bf16* Kt = &Kbuf[cur][0][0];    // [key_local][d], ld=64
        const __bf16* Vl = &Vbuf[cur][0][0];    // [d][key_local], ld=64
#endif
        // ---- scores: S = Q @ K^T ----
        f32x8 s[4];
#pragma unroll
        for (int t = 0; t < 4; ++t) {
            f32x8 c = {};
#if USE_TDM
            c = wmma_bf16(aq0, load_b_tr(Kt, 64, t * 16, 0,  lane), c);
            c = wmma_bf16(aq1, load_b_tr(Kt, 64, t * 16, 32, lane), c);
#else
            c = wmma_bf16(aq0, load_b_tr(Kp, DK, kk + t * 16, 0,  lane), c);
            c = wmma_bf16(aq1, load_b_tr(Kp, DK, kk + t * 16, 32, lane), c);
#endif
            s[t] = c;
        }
        // ---- mask + scale (C layout: row m = kh*8+j, col = key) ----
#pragma unroll
        for (int t = 0; t < 4; ++t)
#pragma unroll
            for (int j = 0; j < 8; ++j) {
                unsigned char mv = Mp[(size_t)(kh * 8 + j) * SEQ + kk + t * 16 + n_];
                s[t][j] = mv ? s[t][j] * scale : -1e30f;
            }
        // ---- online softmax ----
        float alpha[8], rs[8];
#pragma unroll
        for (int j = 0; j < 8; ++j) {
            float tm = red16_max(fmaxf(fmaxf(s[0][j], s[1][j]),
                                       fmaxf(s[2][j], s[3][j])));
            float mn = fmaxf(mrun[j], tm);
            alpha[j] = __expf(mrun[j] - mn);
            mrun[j]  = mn;
        }
#pragma unroll
        for (int t = 0; t < 4; ++t)
#pragma unroll
            for (int j = 0; j < 8; ++j)
                s[t][j] = __expf(s[t][j] - mrun[j]);
#pragma unroll
        for (int j = 0; j < 8; ++j) {
            rs[j] = red16_sum((s[0][j] + s[1][j]) + (s[2][j] + s[3][j]));
            lrun[j] = lrun[j] * alpha[j] + rs[j];
        }
#pragma unroll
        for (int t = 0; t < 4; ++t)
#pragma unroll
            for (int j = 0; j < 8; ++j) acc[t][j] *= alpha[j];
        // ---- P: C layout -> A layout via wave-private LDS strip ----
#pragma unroll
        for (int t = 0; t < 4; ++t)
#pragma unroll
            for (int j = 0; j < 8; ++j)
                Plds[wline][kh * 8 + j][t * 16 + n_] = (__bf16)s[t][j];
        // ---- O += P @ V ----
#pragma unroll
        for (int c = 0; c < 2; ++c) {
            bf16x16 ap = load_a_bf16(&Plds[wline][0][0], 64, c * 32, lane);
#pragma unroll
            for (int t = 0; t < 4; ++t) {
#if USE_TDM
                bf16x16 bv = load_b_tr(Vl, 64, t * 16, c * 32, lane);
#else
                bf16x16 bv = load_b_tr(Vp, SEQ, t * 16, kk + c * 32, lane);
#endif
                acc[t] = wmma_bf16(ap, bv, acc[t]);
            }
        }
#if USE_TDM
        __syncthreads();                        // all done reading before overwrite
#endif
    }

    // ---- epilogue: O = acc / l -> Oc[b][q][h*64+d] (bf16) ----
#pragma unroll
    for (int t = 0; t < 4; ++t)
#pragma unroll
        for (int j = 0; j < 8; ++j) {
            int q = q0 + kh * 8 + j;
            int d = h * 64 + t * 16 + n_;
            Oc[((size_t)b * SEQ + q) * DM + d] = (__bf16)(acc[t][j] / lrun[j]);
        }
}

// ---------------------------------------------------------------------------
// Output projection: out = Oc @ Wo^T + bo (fp32 out), 16x128 per wave.
// ---------------------------------------------------------------------------
__global__ void __launch_bounds__(256)
outproj_kernel(const __bf16* __restrict__ Oc, const __bf16* __restrict__ Wo,
               const float* __restrict__ bias, float* __restrict__ out) {
    int wid  = (blockIdx.x * blockDim.x + threadIdx.x) >> 5;
    int lane = threadIdx.x & 31;
    const int NSTRIP = DM / 128;
    int mt = wid / NSTRIP, nt = wid % NSTRIP;
    int row0 = mt * 16, ncol0 = nt * 128;
    const __bf16* A = Oc + (size_t)row0 * DM;

    f32x8 acc[8] = {};
    for (int k0 = 0; k0 < DM; k0 += 32) {
        if (k0 + 32 < DM)
            __builtin_prefetch(Wo + (size_t)(ncol0 + (lane & 15)) * DM + k0 + 32, 0, 3);
        bf16x16 a = load_a_bf16(A, DM, k0, lane);
#pragma unroll
        for (int t = 0; t < 8; ++t) {
            bf16x16 b = load_b_tr(Wo, DM, ncol0 + t * 16, k0, lane);
            acc[t] = wmma_bf16(a, b, acc[t]);
        }
    }
    int n_ = lane & 15, kh = lane >> 4;
#pragma unroll
    for (int t = 0; t < 8; ++t) {
        int ncol = ncol0 + t * 16 + n_;
        float bv = bias[ncol];
#pragma unroll
        for (int j = 0; j < 8; ++j) {
            int r = row0 + kh * 8 + j;
            out[(size_t)r * DM + ncol] = acc[t][j] + bv;
        }
    }
}

// ---------------------------------------------------------------------------
extern "C" void kernel_launch(void* const* d_in, const int* in_sizes, int n_in,
                              void* d_out, int out_size, void* d_ws, size_t ws_size,
                              hipStream_t stream) {
    const float* q    = (const float*)d_in[0];
    const float* k    = (const float*)d_in[1];
    const float* v    = (const float*)d_in[2];
    const unsigned char* mask = (const unsigned char*)d_in[3];
    const float* Wq = (const float*)d_in[4];
    const float* bq = (const float*)d_in[5];
    const float* Wk = (const float*)d_in[6];
    const float* bk = (const float*)d_in[7];
    const float* Wv = (const float*)d_in[8];
    const float* bv = (const float*)d_in[9];
    const float* Wo = (const float*)d_in[10];
    const float* bo = (const float*)d_in[11];
    float* out = (float*)d_out;

    char* p = (char*)d_ws;
    auto take = [&](size_t bytes) { char* r = p; p += (bytes + 255) & ~(size_t)255; return r; };
    __bf16* Wqb = (__bf16*)take((size_t)DM * DM * 2);
    __bf16* Wkb = (__bf16*)take((size_t)DM * DM * 2);
    __bf16* Wvb = (__bf16*)take((size_t)DM * DM * 2);
    __bf16* Wob = (__bf16*)take((size_t)DM * DM * 2);
    __bf16* Qh  = (__bf16*)take((size_t)BATCH * NH * SEQ * DK * 2);
    __bf16* Kh  = (__bf16*)take((size_t)BATCH * NH * SEQ * DK * 2);
    __bf16* Vtw = (__bf16*)take((size_t)BATCH * NH * DK * SEQ * 2);
    __bf16* Oc  = (__bf16*)take((size_t)BATCH * SEQ * DM * 2);

    const int nW = DM * DM;
    cvt_f32_bf16_kernel<<<nW / 4 / 256, 256, 0, stream>>>(Wq, Wqb, nW);
    cvt_f32_bf16_kernel<<<nW / 4 / 256, 256, 0, stream>>>(Wk, Wkb, nW);
    cvt_f32_bf16_kernel<<<nW / 4 / 256, 256, 0, stream>>>(Wv, Wvb, nW);
    cvt_f32_bf16_kernel<<<nW / 4 / 256, 256, 0, stream>>>(Wo, Wob, nW);

    // 512 m-tiles x 8 n-strips = 4096 waves -> 512 blocks x 8 waves
    proj_kernel<0><<<512, 256, 0, stream>>>(q, Wqb, bq, Qh);
    proj_kernel<0><<<512, 256, 0, stream>>>(k, Wkb, bk, Kh);
    proj_kernel<1><<<512, 256, 0, stream>>>(v, Wvb, bv, Vtw);

    // 8 batch x 16 heads x 16 q-groups = 2048 blocks x 4 waves
    attn_kernel<<<2048, 128, 0, stream>>>(Qh, Kh, Vtw, mask, Oc);

    outproj_kernel<<<512, 256, 0, stream>>>(Oc, Wob, bo, out);
    (void)in_sizes; (void)n_in; (void)out_size; (void)ws_size;
}
